// QuestionDecoder_52948356825457
// MI455X (gfx1250) — matrix operand order repaired and checked by reading, not compiled, not run on hardware
//
#include <hip/hip_runtime.h>

// ---------------------------------------------------------------------------
// Types / helpers
// ---------------------------------------------------------------------------
typedef __bf16 bf16;
typedef __attribute__((ext_vector_type(16))) __bf16 v16bf;
typedef __attribute__((ext_vector_type(8)))  float  v8f;
typedef __attribute__((ext_vector_type(4)))  int    i4;

#define DEVINL __device__ __forceinline__

struct I4x2 { i4 a, b; };

DEVINL bf16 f2bf(float f) {
  unsigned u = __float_as_uint(f);
  unsigned r = (u + 0x7FFFu + ((u >> 16) & 1u)) >> 16;  // round-to-nearest-even
  unsigned short s = (unsigned short)r;
  return __builtin_bit_cast(bf16, s);
}
DEVINL float bf2f(bf16 h) {
  unsigned short s = __builtin_bit_cast(unsigned short, h);
  return __uint_as_float(((unsigned)s) << 16);
}
DEVINL float sigf(float x) { return 1.f / (1.f + __expf(-x)); }

DEVINL v8f v8f_zero() { v8f z = {0.f,0.f,0.f,0.f,0.f,0.f,0.f,0.f}; return z; }

// Monotonic float<->uint mapping for atomicMax-based scatter-max
DEVINL unsigned fenc(float f) {
  unsigned u = __float_as_uint(f);
  return (u & 0x80000000u) ? ~u : (u ^ 0x80000000u);
}
DEVINL float fdec(unsigned e) {
  unsigned u = (e & 0x80000000u) ? (e ^ 0x80000000u) : ~e;
  return __uint_as_float(u);
}

// ---------------------------------------------------------------------------
// WMMA fragment loaders (wave32, v_wmma_f32_16x16x32_bf16)
// All loads are full 16-byte vectors (b128), branch-free.
//
// A 16x32 (MxK) tile from row-major [M,K]; per ISA layout:
//   lanes 0-15: row M=lane,    elems 0..7 = K 0..7,  elems 8..15 = K 16..23
//   lanes16-31: row M=lane-16, elems 0..7 = K 8..15, elems 8..15 = K 24..31
// Requires: (lda*2) % 16 == 0, k0 % 32 == 0, 16B-aligned base.
// ---------------------------------------------------------------------------
DEVINL v16bf load_a_frag(const bf16* __restrict__ A, int lda, int m0, int k0, int lane) {
  int row = m0 + (lane & 15);
  int kg  = (lane >> 4) & 1;
  const bf16* p = A + (long)row * lda + k0 + 8 * kg;
  I4x2 s;
  s.a = *(const i4*)(p);        // K block base+0..7
  s.b = *(const i4*)(p + 16);   // K block base+16..23
  return __builtin_bit_cast(v16bf, s);
}

// B (weights) stored row-major [N,K]; WMMA consumes B as KxN.
// lane n = lane%16, elems 0..7 = K 8*(lane/16)+0..7, elems 8..15 = K 16+8*(lane/16)+0..7.
DEVINL v16bf load_b_frag_nc(const bf16* __restrict__ W, int ldb, int n0, int k0, int lane) {
  int n  = n0 + (lane & 15);
  int kg = (lane >> 4) & 1;
  const bf16* p = W + (long)n * ldb + k0 + kg * 8;
  I4x2 s;
  s.a = *(const i4*)(p);
  s.b = *(const i4*)(p + 16);
  return __builtin_bit_cast(v16bf, s);
}

// Guarded variant for ragged N (e.g. V=30522): clamp row, then AND-mask to
// zero -- branchless, loads stay b128.
DEVINL v16bf load_b_frag(const bf16* __restrict__ W, int ldb, int n0, int k0,
                         int lane, int Nbound) {
  int n  = n0 + (lane & 15);
  int kg = (lane >> 4) & 1;
  bool ok = (n < Nbound);
  int nc = ok ? n : 0;
  const bf16* p = W + (long)nc * ldb + k0 + kg * 8;
  I4x2 s;
  s.a = *(const i4*)(p);
  s.b = *(const i4*)(p + 16);
  int m = ok ? ~0 : 0;
  i4 mv = { m, m, m, m };
  s.a &= mv;
  s.b &= mv;
  return __builtin_bit_cast(v16bf, s);
}

DEVINL v8f wmma_bf16(v16bf a, v16bf b, v8f c) {
  return __builtin_amdgcn_wmma_f32_16x16x32_bf16(false, a, false, b, (short)0, c,
                                                 false, false);
}

// ---------------------------------------------------------------------------
// Generic GEMM: C[M,N] = A[M,K] @ W[N,K]^T (+bias). Optional bf16 output.
// Block = 8 waves = 256 thr. Wave grid 2(M) x 4(N); each wave computes a
// 32 x 64 strip (2 M-tiles x 4 N-tiles = 8 accumulators) so every B fragment
// feeds two WMMAs. Requires M % 64 == 0 (holds for all call sites).
// Wave-uniform fast path skips the ragged-N masking for full column strips.
// Batched via blockIdx.z with element strides sA/sB/sC.
// ---------------------------------------------------------------------------
template <bool GUARD>
DEVINL void gemm_kloop(const bf16* __restrict__ Ab, const bf16* __restrict__ Wb,
                       int K, int lda, int ldb, int m0, int n0, int lane, int N,
                       v8f acc[2][4]) {
  for (int k0 = 0; k0 < K; k0 += 32) {
    v16bf a0 = load_a_frag(Ab, lda, m0, k0, lane);
    v16bf a1 = load_a_frag(Ab, lda, m0 + 16, k0, lane);
#pragma unroll
    for (int j = 0; j < 4; ++j) {
      v16bf b = GUARD ? load_b_frag(Wb, ldb, n0 + j * 16, k0, lane, N)
                      : load_b_frag_nc(Wb, ldb, n0 + j * 16, k0, lane);
      acc[0][j] = wmma_bf16(a0, b, acc[0][j]);
      acc[1][j] = wmma_bf16(a1, b, acc[1][j]);
    }
  }
}

__global__ __launch_bounds__(256)
void gemm_bf16_kernel(const bf16* __restrict__ A, const bf16* __restrict__ W,
                      const float* __restrict__ bias,
                      float* __restrict__ Cf, bf16* __restrict__ Cb,
                      int M, int N, int K, int lda, int ldb, int ldc,
                      long sA, long sB, long sC) {
  int batch = blockIdx.z;
  const bf16* Ab = A + (long)batch * sA;
  const bf16* Wb = W + (long)batch * sB;

  int lane = threadIdx.x & 31;
  int wave = threadIdx.x >> 5;
  int wm = wave >> 2;      // 0..1
  int wn = wave & 3;       // 0..3
  int m0 = blockIdx.y * 64 + wm * 32;
  int n0 = blockIdx.x * 256 + wn * 64;
  if (m0 >= M) return;

  v8f acc[2][4];
#pragma unroll
  for (int t = 0; t < 2; ++t)
#pragma unroll
    for (int j = 0; j < 4; ++j) acc[t][j] = v8f_zero();

  if (n0 + 64 <= N) {
    gemm_kloop<false>(Ab, Wb, K, lda, ldb, m0, n0, lane, N, acc);
  } else {
    gemm_kloop<true>(Ab, Wb, K, lda, ldb, m0, n0, lane, N, acc);
  }

  int nl = lane & 15;
#pragma unroll
  for (int t = 0; t < 2; ++t) {
    int mrow = m0 + t * 16 + ((lane >> 4) << 3);
#pragma unroll
    for (int j = 0; j < 4; ++j) {
      int n = n0 + j * 16 + nl;
      if (n >= N) continue;
      float bv = bias ? bias[n] : 0.f;
      if (Cb) {
        bf16* p = Cb + (long)batch * sC;
#pragma unroll
        for (int r = 0; r < 8; ++r)
          p[(long)(mrow + r) * ldc + n] = f2bf(acc[t][j][r] + bv);
      } else {
        float* p = Cf + (long)batch * sC;
#pragma unroll
        for (int r = 0; r < 8; ++r)
          p[(long)(mrow + r) * ldc + n] = acc[t][j][r] + bv;
      }
    }
  }
}

// ---------------------------------------------------------------------------
// Persistent LSTM scan. Batch=16 rows == one WMMA M-tile.
//   gates g[16, 4*HCT] = xp[:, t, :] + h @ Whh^T     (torch gate order i,f,g,o)
// 16 waves; each wave owns NB hidden 16-col blocks across ALL FOUR gates, so
// i/f/g/o of a given (m, j) live in the same lane -> cell update is local.
// h kept in LDS (bf16), c in registers across the entire scan.
// grid.x = 2 selects fwd/bwd parameter set (context BiLSTM); 1 for decoder.
// ---------------------------------------------------------------------------
template <int HCT, int NB>
__global__ __launch_bounds__(512)
void lstm_scan_kernel(const float* __restrict__ xp0, const bf16* __restrict__ Whh0,
                      const float* __restrict__ xp1, const bf16* __restrict__ Whh1,
                      const float* __restrict__ h0, const float* __restrict__ c0,
                      bf16* __restrict__ out, int outLd,
                      bf16* __restrict__ outT, int HT, int T) {
  const int dir = blockIdx.x;                 // 0 = forward, 1 = backward
  const float* xp  = dir ? xp1 : xp0;
  const bf16* Whh  = dir ? Whh1 : Whh0;
  const int outOff = dir * HCT;
  const int G4 = 4 * HCT;

  __shared__ bf16 hbuf[16 * HCT];

  int lane = threadIdx.x & 31;
  int wave = threadIdx.x >> 5;                // 0..15
  int nl = lane & 15;
  int mb = (lane >> 4) << 3;                  // 0 or 8

  for (int i = threadIdx.x; i < 16 * HCT; i += blockDim.x)
    hbuf[i] = h0 ? f2bf(h0[i]) : f2bf(0.f);

  float creg[NB][8];
#pragma unroll
  for (int jb = 0; jb < NB; ++jb) {
    int j = (wave * NB + jb) * 16 + nl;
#pragma unroll
    for (int r = 0; r < 8; ++r)
      creg[jb][r] = c0 ? c0[(mb + r) * HCT + j] : 0.f;
  }
  __syncthreads();

  for (int t = 0; t < T; ++t) {
    int tt = dir ? (T - 1 - t) : t;

    v8f acc[NB][4];
#pragma unroll
    for (int jb = 0; jb < NB; ++jb)
#pragma unroll
      for (int g = 0; g < 4; ++g) acc[jb][g] = v8f_zero();

    for (int k0 = 0; k0 < HCT; k0 += 32) {
      v16bf a = load_a_frag(hbuf, HCT, 0, k0, lane);   // h as A matrix (LDS)
#pragma unroll
      for (int jb = 0; jb < NB; ++jb) {
        int nblk = (wave * NB + jb) * 16;
#pragma unroll
        for (int g = 0; g < 4; ++g) {
          v16bf b = load_b_frag_nc(Whh, HCT, g * HCT + nblk, k0, lane);
          acc[jb][g] = wmma_bf16(a, b, acc[jb][g]);
        }
      }
    }
    __syncthreads();  // all waves done reading hbuf

#pragma unroll
    for (int jb = 0; jb < NB; ++jb) {
      int j = (wave * NB + jb) * 16 + nl;
#pragma unroll
      for (int r = 0; r < 8; ++r) {
        int m = mb + r;
        long row = (long)(m * T + tt) * G4;
        float xi = acc[jb][0][r] + xp[row + j];
        float xf = acc[jb][1][r] + xp[row + HCT + j];
        float xg = acc[jb][2][r] + xp[row + 2 * HCT + j];
        float xo = acc[jb][3][r] + xp[row + 3 * HCT + j];
        float c = sigf(xf) * creg[jb][r] + sigf(xi) * tanhf(xg);
        creg[jb][r] = c;
        float hval = sigf(xo) * tanhf(c);
        bf16 hb = f2bf(hval);
        hbuf[m * HCT + j] = hb;
        out[(long)(m * T + tt) * outLd + outOff + j] = hb;
        if (outT)
          outT[((long)m * HT + outOff + j) * T + tt] = hb;
      }
    }
    __syncthreads();
  }
}

// ---------------------------------------------------------------------------
// Glue kernels
// ---------------------------------------------------------------------------
__global__ void f2bf_kernel(const float* __restrict__ in, bf16* __restrict__ out, long n) {
  long i = (long)blockIdx.x * blockDim.x + threadIdx.x;
  if (i < n) out[i] = f2bf(in[i]);
}

// out[c*R + r] = in[r*C + c]  (transpose + convert; for Wa^T)
__global__ void transpose_f2bf_kernel(const float* __restrict__ in, bf16* __restrict__ out,
                                      int R, int C) {
  long i = (long)blockIdx.x * blockDim.x + threadIdx.x;
  if (i >= (long)R * C) return;
  int r = (int)(i / C), c = (int)(i % C);
  out[(long)c * R + r] = f2bf(in[i]);
}

__global__ void embed_ctx_kernel(const int* __restrict__ c_ids, const int* __restrict__ amask,
                                 const float* __restrict__ wemb, const float* __restrict__ aemb,
                                 bf16* __restrict__ out, long n, int D) {
  long i = (long)blockIdx.x * blockDim.x + threadIdx.x;
  if (i >= n) return;
  long bt = i / D; int d = (int)(i % D);
  out[i] = f2bf(wemb[(long)c_ids[bt] * D + d] + aemb[(long)amask[bt] * D + d]);
}

__global__ void embed_q_kernel(const int* __restrict__ q_ids, const float* __restrict__ wemb,
                               bf16* __restrict__ out, long n, int D) {
  long i = (long)blockIdx.x * blockDim.x + threadIdx.x;
  if (i >= n) return;
  long bt = i / D; int d = (int)(i % D);
  out[i] = f2bf(wemb[(long)q_ids[bt] * D + d]);
}

// h0 = zq @ Wh^T + bh ; c0 = zq @ Wc^T + bc   (16x512, K=64; tiny)
__global__ void h0c0_kernel(const float* __restrict__ zq,
                            const float* __restrict__ Wh, const float* __restrict__ bh,
                            const float* __restrict__ Wc, const float* __restrict__ bc,
                            float* __restrict__ h0, float* __restrict__ c0) {
  int i = blockIdx.x * blockDim.x + threadIdx.x;
  if (i >= 16 * 512) return;
  int m = i / 512, j = i % 512;
  float s1 = bh[j], s2 = bc[j];
#pragma unroll 4
  for (int k = 0; k < 64; ++k) {
    float z = zq[m * 64 + k];
    s1 += z * Wh[j * 64 + k];
    s2 += z * Wc[j * 64 + k];
  }
  h0[i] = s1;
  c0[i] = s2;
}

// Mask + softmax over one (b,q) row of LC=384; logits buffer updated in place
// (becomes attn_logits), probabilities written as bf16.
__global__ __launch_bounds__(256)
void mask_softmax_kernel(const int* __restrict__ c_ids, const int* __restrict__ q_ids,
                         float* __restrict__ logits, bf16* __restrict__ attn) {
  const int LC = 384, LQ = 64;
  int row = blockIdx.x;
  int b = row / LQ, q = row % LQ;
  bool qm = q_ids[b * LQ + q] != 0;
  float* L = logits + (long)row * LC;
  __shared__ float red[256];

  float mx = -3.0e38f;
  for (int c = threadIdx.x; c < LC; c += 256) {
    bool cm = qm && (c_ids[b * LC + c] != 0);
    float v = cm ? L[c] : -1.0e4f;
    L[c] = v;
    mx = fmaxf(mx, v);
  }
  red[threadIdx.x] = mx; __syncthreads();
  for (int s = 128; s > 0; s >>= 1) {
    if (threadIdx.x < s) red[threadIdx.x] = fmaxf(red[threadIdx.x], red[threadIdx.x + s]);
    __syncthreads();
  }
  mx = red[0]; __syncthreads();

  float sum = 0.f;
  for (int c = threadIdx.x; c < LC; c += 256) sum += __expf(L[c] - mx);
  red[threadIdx.x] = sum; __syncthreads();
  for (int s = 128; s > 0; s >>= 1) {
    if (threadIdx.x < s) red[threadIdx.x] += red[threadIdx.x + s];
    __syncthreads();
  }
  float inv = 1.f / red[0];
  for (int c = threadIdx.x; c < LC; c += 256)
    attn[(long)row * LC + c] = f2bf(__expf(L[c] - mx) * inv);
}

// maxout over pairs: out[row, d] = max(h2[row, 2d], h2[row, 2d+1]) as bf16
__global__ void maxout_kernel(const float* __restrict__ h2, bf16* __restrict__ out, long n, int D) {
  long i = (long)blockIdx.x * blockDim.x + threadIdx.x;
  if (i >= n) return;
  long row = i / D; int d = (int)(i % D);
  const float* p = h2 + row * (2 * D) + 2 * d;
  out[i] = f2bf(fmaxf(p[0], p[1]));
}

__global__ void copy_init_kernel(unsigned* __restrict__ buf, long n, unsigned val) {
  long i = (long)blockIdx.x * blockDim.x + threadIdx.x;
  if (i < n) buf[i] = val;
}

// scatter-max attn_logits into vocab buckets by context token id
__global__ void copy_scatter_kernel(const int* __restrict__ c_ids,
                                    const float* __restrict__ logits,
                                    unsigned* __restrict__ buf) {
  const int LC = 384, LQ = 64, V = 30522;
  long i = (long)blockIdx.x * blockDim.x + threadIdx.x;
  if (i >= (long)16 * LQ * LC) return;
  long row = i / LC; int c = (int)(i % LC);
  int b = (int)(row / LQ);
  int v = c_ids[b * LC + c];
  atomicMax(&buf[row * V + v], fenc(logits[i]));
}

__global__ void copy_add_kernel(const unsigned* __restrict__ buf, float* __restrict__ out,
                                long n, unsigned negEnc) {
  long i = (long)blockIdx.x * blockDim.x + threadIdx.x;
  if (i >= n) return;
  unsigned e = buf[i];
  out[i] += (e != negEnc) ? fdec(e) : 0.f;
}

// ---------------------------------------------------------------------------
// Host side
// ---------------------------------------------------------------------------
static void launch_gemm(const bf16* A, const bf16* W, const float* bias,
                        float* Cf, bf16* Cb, int M, int N, int K,
                        int lda, int ldb, int ldc,
                        long sA, long sB, long sC, int batches, hipStream_t s) {
  dim3 g((N + 255) / 256, (M + 63) / 64, batches);
  gemm_bf16_kernel<<<g, 256, 0, s>>>(A, W, bias, Cf, Cb, M, N, K, lda, ldb, ldc, sA, sB, sC);
}

extern "C" void kernel_launch(void* const* d_in, const int* in_sizes, int n_in,
                              void* d_out, int out_size, void* d_ws, size_t ws_size,
                              hipStream_t stream) {
  const int B = 16, LC = 384, LQ = 64, V = 30522, D = 768, H = 512, HC = 256;

  const int*   c_ids  = (const int*)  d_in[0];
  const int*   q_ids  = (const int*)  d_in[1];
  const int*   amask  = (const int*)  d_in[2];
  const float* zq     = (const float*)d_in[3];
  const float* wemb   = (const float*)d_in[4];
  const float* aemb   = (const float*)d_in[5];
  const float* cWih_f = (const float*)d_in[6];
  const float* cWhh_f = (const float*)d_in[7];
  const float* cb_f   = (const float*)d_in[8];
  const float* cWih_b = (const float*)d_in[9];
  const float* cWhh_b = (const float*)d_in[10];
  const float* cb_b   = (const float*)d_in[11];
  const float* qWih   = (const float*)d_in[12];
  const float* qWhh   = (const float*)d_in[13];
  const float* qb     = (const float*)d_in[14];
  const float* Wh     = (const float*)d_in[15];
  const float* bh     = (const float*)d_in[16];
  const float* Wc     = (const float*)d_in[17];
  const float* bc     = (const float*)d_in[18];
  const float* Wa     = (const float*)d_in[19];
  const float* W1     = (const float*)d_in[20];
  const float* b1     = (const float*)d_in[21];
  const float* W2     = (const float*)d_in[22];
  const float* b2     = (const float*)d_in[23];

  // Deterministic workspace arena
  size_t cur = 0;
  auto carve = [&](size_t bytes) -> void* {
    void* p = (char*)d_ws + cur;
    cur += (bytes + 255) & ~(size_t)255;
    return p;
  };
  bf16*  c_emb   = (bf16*)carve((size_t)B * LC * D * 2);
  bf16*  q_emb   = (bf16*)carve((size_t)B * LQ * D * 2);
  bf16*  cWihF_b = (bf16*)carve((size_t)4 * HC * D * 2);
  bf16*  cWhhF_b = (bf16*)carve((size_t)4 * HC * HC * 2);
  bf16*  cWihB_b = (bf16*)carve((size_t)4 * HC * D * 2);
  bf16*  cWhhB_b = (bf16*)carve((size_t)4 * HC * HC * 2);
  bf16*  qWih_b  = (bf16*)carve((size_t)4 * H * D * 2);
  bf16*  qWhh_b  = (bf16*)carve((size_t)4 * H * H * 2);
  bf16*  WaT_b   = (bf16*)carve((size_t)H * H * 2);
  bf16*  W1_b    = (bf16*)carve((size_t)(2 * H) * (2 * H) * 2);
  bf16*  W2_b    = (bf16*)carve((size_t)(2 * D) * (2 * H) * 2);
  bf16*  wemb_b  = (bf16*)carve((size_t)V * D * 2);
  float* xpf     = (float*)carve((size_t)B * LC * 4 * HC * 4);
  float* xpb     = (float*)carve((size_t)B * LC * 4 * HC * 4);
  float* xq      = (float*)carve((size_t)B * LQ * 4 * H * 4);
  float* h0      = (float*)carve((size_t)B * H * 4);
  float* c0      = (float*)carve((size_t)B * H * 4);
  bf16*  cout_b  = (bf16*)carve((size_t)B * LC * H * 2);
  bf16*  coutT_b = (bf16*)carve((size_t)B * H * LC * 2);
  bf16*  hcat    = (bf16*)carve((size_t)B * LQ * 2 * H * 2);
  bf16*  qWa_b   = (bf16*)carve((size_t)B * LQ * H * 2);
  float* logits  = (float*)carve((size_t)B * LQ * LC * 4);
  bf16*  attn_b  = (bf16*)carve((size_t)B * LQ * LC * 2);
  bf16*  h1_b    = (bf16*)carve((size_t)B * LQ * 2 * H * 2);
  float* h2      = (float*)carve((size_t)B * LQ * 2 * D * 4);
  bf16*  mo_b    = (bf16*)carve((size_t)B * LQ * D * 2);
  unsigned* cpy  = (unsigned*)carve((size_t)B * LQ * V * 4);

  auto cvt = [&](const float* in, bf16* out, long n) {
    f2bf_kernel<<<(unsigned)((n + 255) / 256), 256, 0, stream>>>(in, out, n);
  };

  // --- weight conversions to bf16 ---
  cvt(cWih_f, cWihF_b, (long)4 * HC * D);
  cvt(cWhh_f, cWhhF_b, (long)4 * HC * HC);
  cvt(cWih_b, cWihB_b, (long)4 * HC * D);
  cvt(cWhh_b, cWhhB_b, (long)4 * HC * HC);
  cvt(qWih, qWih_b, (long)4 * H * D);
  cvt(qWhh, qWhh_b, (long)4 * H * H);
  cvt(W1, W1_b, (long)(2 * H) * (2 * H));
  cvt(W2, W2_b, (long)(2 * D) * (2 * H));
  cvt(wemb, wemb_b, (long)V * D);
  transpose_f2bf_kernel<<<(H * H + 255) / 256, 256, 0, stream>>>(Wa, WaT_b, H, H);

  // --- embeddings ---
  {
    long n = (long)B * LC * D;
    embed_ctx_kernel<<<(unsigned)((n + 255) / 256), 256, 0, stream>>>(c_ids, amask, wemb, aemb, c_emb, n, D);
    long nq = (long)B * LQ * D;
    embed_q_kernel<<<(unsigned)((nq + 255) / 256), 256, 0, stream>>>(q_ids, wemb, q_emb, nq, D);
  }

  // --- input projections (parallel-over-time GEMMs) ---
  launch_gemm(c_emb, cWihF_b, cb_f, xpf, nullptr, B * LC, 4 * HC, D, D, D, 4 * HC, 0, 0, 0, 1, stream);
  launch_gemm(c_emb, cWihB_b, cb_b, xpb, nullptr, B * LC, 4 * HC, D, D, D, 4 * HC, 0, 0, 0, 1, stream);
  launch_gemm(q_emb, qWih_b, qb, xq, nullptr, B * LQ, 4 * H, D, D, D, 4 * H, 0, 0, 0, 1, stream);

  // --- decoder init state ---
  h0c0_kernel<<<(16 * 512 + 255) / 256, 256, 0, stream>>>(zq, Wh, bh, Wc, bc, h0, c0);

  // --- context BiLSTM scan (grid.x = 2 directions, persistent workgroups) ---
  lstm_scan_kernel<256, 1><<<2, 512, 0, stream>>>(
      xpf, cWhhF_b, xpb, cWhhB_b, nullptr, nullptr,
      cout_b, H, coutT_b, H, LC);

  // --- decoder LSTM scan (writes q_out into hcat[:, 0:512]) ---
  lstm_scan_kernel<512, 2><<<1, 512, 0, stream>>>(
      xq, qWhh_b, xq, qWhh_b, h0, c0,
      hcat, 2 * H, nullptr, 0, LQ);

  // --- attention: qWa = q_out @ Wa (via Wa^T as [N=k, K=h]) ---
  launch_gemm(hcat, WaT_b, nullptr, nullptr, qWa_b, B * LQ, H, H, 2 * H, H, H, 0, 0, 0, 1, stream);

  // scores[b] = qWa[b] @ c_out[b]^T  (c_out already [c, h] = [N, K])
  launch_gemm(qWa_b, cout_b, nullptr, logits, nullptr, LQ, LC, H, H, H, LC,
              (long)LQ * H, (long)LC * H, (long)LQ * LC, B, stream);

  // mask + softmax (logits -> attn_logits in place; attn as bf16)
  mask_softmax_kernel<<<B * LQ, 256, 0, stream>>>(c_ids, q_ids, logits, attn_b);

  // ctx[b] = attn[b] @ c_out[b]  (needs c_out^T as [N=h, K=c]) -> hcat[:, 512:]
  launch_gemm(attn_b, coutT_b, nullptr, nullptr, hcat + H, LQ, H, LC, LC, LC, 2 * H,
              (long)LQ * LC, (long)H * LC, (long)LQ * 2 * H, B, stream);

  // --- output head ---
  launch_gemm(hcat, W1_b, b1, nullptr, h1_b, B * LQ, 2 * H, 2 * H, 2 * H, 2 * H, 2 * H,
              0, 0, 0, 1, stream);
  launch_gemm(h1_b, W2_b, b2, h2, nullptr, B * LQ, 2 * D, 2 * H, 2 * H, 2 * H, 2 * D,
              0, 0, 0, 1, stream);
  {
    long n = (long)B * LQ * D;
    maxout_kernel<<<(unsigned)((n + 255) / 256), 256, 0, stream>>>(h2, mo_b, n, D);
  }
  // gen = maxout @ word_emb^T  -> d_out (f32), ragged N = 30522 handled in kernel
  launch_gemm(mo_b, wemb_b, nullptr, (float*)d_out, nullptr, B * LQ, V, D, D, D, V,
              0, 0, 0, 1, stream);

  // --- copy mechanism: scatter-max attn_logits into vocab, add into d_out ---
  {
    long nv = (long)B * LQ * V;
    union { float f; unsigned u; } cv; cv.f = -30000.0f;
    unsigned negE = (cv.u & 0x80000000u) ? ~cv.u : (cv.u ^ 0x80000000u);
    copy_init_kernel<<<(unsigned)((nv + 255) / 256), 256, 0, stream>>>(cpy, nv, negE);
    long ns = (long)B * LQ * LC;
    copy_scatter_kernel<<<(unsigned)((ns + 255) / 256), 256, 0, stream>>>(c_ids, logits, cpy);
    copy_add_kernel<<<(unsigned)((nv + 255) / 256), 256, 0, stream>>>(cpy, (float*)d_out, nv, negE);
  }
}
